// MultiLatentAttention_67611375173816
// MI455X (gfx1250) — matrix operand order repaired and checked by reading, not compiled, not run on hardware
//
#include <hip/hip_runtime.h>

// ---------------------------------------------------------------------------
// Types
// ---------------------------------------------------------------------------
typedef __attribute__((ext_vector_type(16))) __bf16 v16bf;
typedef __attribute__((ext_vector_type(8)))  __bf16 v8bf;
typedef __attribute__((ext_vector_type(8)))  float  v8f;
typedef __attribute__((ext_vector_type(4)))  unsigned short v4u16;

// Problem constants (match setup_inputs)
#define BB   4
#define SS   2048
#define DIM  4096
#define KVD  512
#define NH   32
#define HD   128
#define MM   (BB * SS)   // 8192 rows

// ---------------------------------------------------------------------------
// f32 -> bf16 round-to-nearest-even
// ---------------------------------------------------------------------------
__device__ __forceinline__ unsigned short f2bf(float f) {
    unsigned u = __float_as_uint(f);
    u += 0x7FFFu + ((u >> 16) & 1u);
    return (unsigned short)(u >> 16);
}

__global__ void cvt_f32_bf16(const float* __restrict__ in,
                             unsigned short* __restrict__ out, long n) {
    long i = (long)blockIdx.x * blockDim.x + threadIdx.x;
    long stride = (long)gridDim.x * blockDim.x;
    for (; i < n; i += stride) out[i] = f2bf(in[i]);
}

// ---------------------------------------------------------------------------
// WMMA fragment loaders (contiguous k = 16h + s order in both A and B, wave32)
//
// A (16x32, MxK, 16-bit): lane l<16 -> row M=l, elems[0..15]=K0..15 ;
//   lane l>=16 -> row M=l-16, K16..31 (same k order as B).
// B (32x16, KxN, 16-bit), sourced from row-major W (N,K): lane n<16 ->
//   col N=n = W row n, K0..15 contiguous ; lane n>=16 -> K16..31.
// ---------------------------------------------------------------------------
__device__ __forceinline__ v16bf load_a_frag(const __bf16* __restrict__ A,
                                             int K, int row_base, int k0, int lane) {
    int r = lane & 15, h = lane >> 4;
    const __bf16* p = A + (size_t)(row_base + r) * K + k0 + h * 16;
    v8bf lo = *(const v8bf*)(p);
    v8bf hi = *(const v8bf*)(p + 8);
    v16bf out;
#pragma unroll
    for (int i = 0; i < 8; ++i) { out[i] = lo[i]; out[i + 8] = hi[i]; }
    return out;
}

__device__ __forceinline__ v16bf load_b_frag(const __bf16* __restrict__ W,
                                             int K, int col_base, int k0, int lane) {
    int r = lane & 15, h = lane >> 4;
    const __bf16* p = W + (size_t)(col_base + r) * K + k0 + h * 16;
    return *(const v16bf*)(p);
}

// ---------------------------------------------------------------------------
// GEMM: out[M,N] = A[M,K](bf16) * W[N,K](bf16)^T + bias[N]
// Workgroup: 256 thr = 8 waves (2 in M x 4 in N), tile 128(M) x 256(N).
// Wave tile: 64x64 = 4x4 accumulators of 16x16, K stepped by 32 with
// v_wmma_f32_16x16x32_bf16 -> 16 independent WMMAs per K-step vs 16 b128
// loads (B fragments loaded once per K-step, A streamed per accumulator row
// to bound register liveness: 128 acc + 32 B + 8 A VGPRs).
// OUT_BF16: 1 -> write bf16 (unsigned short), 0 -> write f32.
// ---------------------------------------------------------------------------
template <int OUT_BF16>
__global__ __launch_bounds__(256)
void gemm_bf16_tn(const unsigned short* __restrict__ Au,
                  const unsigned short* __restrict__ Wu,
                  const float* __restrict__ bias,
                  void* __restrict__ out,
                  int M, int N, int K) {
    const __bf16* A = (const __bf16*)Au;
    const __bf16* W = (const __bf16*)Wu;

    const int lane = threadIdx.x & 31;
    const int wave = threadIdx.x >> 5;
    const int wm   = wave >> 2;          // 0..1
    const int wn   = wave & 3;           // 0..3

    const int m0 = blockIdx.y * 128 + wm * 64;   // wave's first output row
    const int n0 = blockIdx.x * 256 + wn * 64;   // wave's first output col

    v8f acc[4][4];
#pragma unroll
    for (int mi = 0; mi < 4; ++mi)
#pragma unroll
        for (int ni = 0; ni < 4; ++ni) acc[mi][ni] = (v8f)(0.0f);

    for (int k0 = 0; k0 < K; k0 += 32) {
        v16bf b[4];
#pragma unroll
        for (int ni = 0; ni < 4; ++ni)
            b[ni] = load_b_frag(W, K, n0 + ni * 16, k0, lane);
#pragma unroll
        for (int mi = 0; mi < 4; ++mi) {
            const v16bf a = load_a_frag(A, K, m0 + mi * 16, k0, lane);
#pragma unroll
            for (int ni = 0; ni < 4; ++ni)
                acc[mi][ni] = __builtin_amdgcn_wmma_f32_16x16x32_bf16(
                    false, a, false, b[ni], (short)0, acc[mi][ni],
                    false, false);
        }
    }

    // Epilogue. C/D layout: elem g of lane l -> M = g + 8*(l>>4), N = l&15.
    const int r = lane & 15, h = lane >> 4;
#pragma unroll
    for (int mi = 0; mi < 4; ++mi) {
#pragma unroll
        for (int ni = 0; ni < 4; ++ni) {
            const int col = n0 + ni * 16 + r;
            const float bv = bias[col];
#pragma unroll
            for (int g = 0; g < 8; ++g) {
                const int row = m0 + mi * 16 + 8 * h + g;
                const float v = acc[mi][ni][g] + bv;
                if (OUT_BF16)
                    ((unsigned short*)out)[(size_t)row * N + col] = f2bf(v);
                else
                    ((float*)out)[(size_t)row * N + col] = v;
            }
        }
    }
}

// ---------------------------------------------------------------------------
// Fused RoPE + elementwise-softmax(head dim) + *v  ->  attn (bf16)
// One wave per (row, head): 128 contiguous f32, 4 per lane.
// RoPE pairs (2j, 2j+1) are lane-local (lane handles d = 4*lane..4*lane+3).
// ---------------------------------------------------------------------------
__global__ __launch_bounds__(256)
void rope_softmax_v(const float* __restrict__ qf,
                    const float* __restrict__ kf,
                    const float* __restrict__ vf,
                    const float* __restrict__ cosT,   // (S, HD/2)
                    const float* __restrict__ sinT,
                    unsigned short* __restrict__ attn) {
    const int lane = threadIdx.x & 31;
    const int wave = threadIdx.x >> 5;
    const long task = (long)blockIdx.x * 8 + wave;   // MM*NH tasks
    const int row  = (int)(task >> 5);               // 0..MM-1
    const int head = (int)(task & 31);
    const int s    = row & (SS - 1);                 // position within sequence

    const size_t base = (size_t)row * DIM + head * HD;
    const int d0 = lane * 4;
    const int j0 = d0 >> 1;                          // rope pair indices j0, j0+1

    const float4 q = *(const float4*)(qf + base + d0);
    const float4 k = *(const float4*)(kf + base + d0);
    const float4 v = *(const float4*)(vf + base + d0);

    const float c0 = cosT[(size_t)s * (HD / 2) + j0];
    const float s0 = sinT[(size_t)s * (HD / 2) + j0];
    const float c1 = cosT[(size_t)s * (HD / 2) + j0 + 1];
    const float s1 = sinT[(size_t)s * (HD / 2) + j0 + 1];

    const float qr0 = q.x * c0 - q.y * s0, qi0 = q.x * s0 + q.y * c0;
    const float qr1 = q.z * c1 - q.w * s1, qi1 = q.z * s1 + q.w * c1;
    const float kr0 = k.x * c0 - k.y * s0, ki0 = k.x * s0 + k.y * c0;
    const float kr1 = k.z * c1 - k.w * s1, ki1 = k.z * s1 + k.w * c1;

    const float scale = 0.015625f;                   // DIM^-0.5 = 1/64
    float t[4];
    t[0] = qr0 * kr0 * scale;
    t[1] = qi0 * ki0 * scale;
    t[2] = qr1 * kr1 * scale;
    t[3] = qi1 * ki1 * scale;

    // wave32 max-reduction
    float m = fmaxf(fmaxf(t[0], t[1]), fmaxf(t[2], t[3]));
#pragma unroll
    for (int off = 16; off > 0; off >>= 1)
        m = fmaxf(m, __shfl_xor(m, off, 32));

    float e[4], sum = 0.f;
#pragma unroll
    for (int i = 0; i < 4; ++i) { e[i] = __expf(t[i] - m); sum += e[i]; }
#pragma unroll
    for (int off = 16; off > 0; off >>= 1)
        sum += __shfl_xor(sum, off, 32);

    const float inv = 1.0f / sum;
    v4u16 pk;
    pk[0] = f2bf(e[0] * inv * v.x);
    pk[1] = f2bf(e[1] * inv * v.y);
    pk[2] = f2bf(e[2] * inv * v.z);
    pk[3] = f2bf(e[3] * inv * v.w);
    *(v4u16*)(attn + base + d0) = pk;
}

// ---------------------------------------------------------------------------
// Host launcher
// ---------------------------------------------------------------------------
extern "C" void kernel_launch(void* const* d_in, const int* in_sizes, int n_in,
                              void* d_out, int out_size, void* d_ws, size_t ws_size,
                              hipStream_t stream) {
    // Inputs (setup_inputs order):
    const float* x        = (const float*)d_in[0];
    // d_in[1] = start_pos (0, unused by the reference math)
    const float* rope_cos = (const float*)d_in[2];
    const float* rope_sin = (const float*)d_in[3];
    const float* W_kvd    = (const float*)d_in[4];
    const float* b_kvd    = (const float*)d_in[5];
    const float* Wq       = (const float*)d_in[6];
    const float* bq       = (const float*)d_in[7];
    const float* W_uk     = (const float*)d_in[8];
    const float* b_uk     = (const float*)d_in[9];
    const float* W_uv     = (const float*)d_in[10];
    const float* b_uv     = (const float*)d_in[11];
    const float* Wo       = (const float*)d_in[12];
    const float* bo       = (const float*)d_in[13];

    // Workspace carve-up
    char* w = (char*)d_ws;
    auto carve = [&](size_t bytes) -> void* {
        void* p = (void*)w;
        w += (bytes + 255) & ~(size_t)255;
        return p;
    };
    unsigned short* xb    = (unsigned short*)carve((size_t)MM  * DIM * 2);
    unsigned short* wqb   = (unsigned short*)carve((size_t)DIM * DIM * 2);
    unsigned short* wkvdb = (unsigned short*)carve((size_t)KVD * DIM * 2);
    unsigned short* wukb  = (unsigned short*)carve((size_t)DIM * KVD * 2);
    unsigned short* wuvb  = (unsigned short*)carve((size_t)DIM * KVD * 2);
    unsigned short* wob   = (unsigned short*)carve((size_t)DIM * DIM * 2);
    unsigned short* lkvb  = (unsigned short*)carve((size_t)MM  * KVD * 2);
    float*          qf    = (float*)carve((size_t)MM * DIM * 4);
    float*          kf    = (float*)carve((size_t)MM * DIM * 4);
    float*          vf    = (float*)carve((size_t)MM * DIM * 4);
    unsigned short* attnb = (unsigned short*)carve((size_t)MM * DIM * 2);

    // 1) f32 -> bf16 conversions
    const int CT = 256, CG = 4096;
    cvt_f32_bf16<<<CG, CT, 0, stream>>>(x,     xb,    (long)MM  * DIM);
    cvt_f32_bf16<<<CG, CT, 0, stream>>>(Wq,    wqb,   (long)DIM * DIM);
    cvt_f32_bf16<<<CG, CT, 0, stream>>>(W_kvd, wkvdb, (long)KVD * DIM);
    cvt_f32_bf16<<<CG, CT, 0, stream>>>(W_uk,  wukb,  (long)DIM * KVD);
    cvt_f32_bf16<<<CG, CT, 0, stream>>>(W_uv,  wuvb,  (long)DIM * KVD);
    cvt_f32_bf16<<<CG, CT, 0, stream>>>(Wo,    wob,   (long)DIM * DIM);

    // 2) GEMMs (grid: N/256 x M/128, 256 threads)
    dim3 blk(256);
    // L_kv (bf16) = x * W_kv_down^T + b
    gemm_bf16_tn<1><<<dim3(KVD / 256, MM / 128), blk, 0, stream>>>(
        xb, wkvdb, b_kvd, (void*)lkvb, MM, KVD, DIM);
    // q (f32) = x * Wq^T + bq
    gemm_bf16_tn<0><<<dim3(DIM / 256, MM / 128), blk, 0, stream>>>(
        xb, wqb, bq, (void*)qf, MM, DIM, DIM);
    // k (f32) = L_kv * W_uk^T + b_uk
    gemm_bf16_tn<0><<<dim3(DIM / 256, MM / 128), blk, 0, stream>>>(
        lkvb, wukb, b_uk, (void*)kf, MM, DIM, KVD);
    // v (f32) = L_kv * W_uv^T + b_uv
    gemm_bf16_tn<0><<<dim3(DIM / 256, MM / 128), blk, 0, stream>>>(
        lkvb, wuvb, b_uv, (void*)vf, MM, DIM, KVD);

    // 3) fused RoPE + softmax(head) + *v -> attn bf16
    const long tasks = (long)MM * NH;                 // 262144 waves
    rope_softmax_v<<<(unsigned)(tasks / 8), 256, 0, stream>>>(
        qf, kf, vf, rope_cos, rope_sin, attnb);

    // 4) out (f32) = attn * Wo^T + bo
    gemm_bf16_tn<0><<<dim3(DIM / 256, MM / 128), blk, 0, stream>>>(
        attnb, wob, bo, d_out, MM, DIM, DIM);
}
